// GATNet_35485019799828
// MI455X (gfx1250) — matrix operand order repaired and checked by reading, not compiled, not run on hardware
//
#include <hip/hip_runtime.h>
#include <math.h>

#define HID 64
#define NEG_SLOPE 0.2f

typedef __attribute__((ext_vector_type(16))) __bf16 v16bf;
typedef __attribute__((ext_vector_type(8)))  float  v8f;

__device__ __forceinline__ unsigned short f2bf(float f) {
  union { float f; unsigned int u; } x; x.f = f;
  unsigned int r = x.u + 0x7FFFu + ((x.u >> 16) & 1u);   // RNE
  return (unsigned short)(r >> 16);
}

// ---- prep: Bt[n*K + k] = bf16(W[k*N + n]), zero-pad n >= N  (N x K col-major)
__global__ void conv_wt_kernel(const float* __restrict__ W,
                               unsigned short* __restrict__ Bt,
                               int K, int N, int Npad)
{
  int i = blockIdx.x * blockDim.x + threadIdx.x;
  if (i >= Npad * K) return;
  int n = i / K, k = i - n * K;
  Bt[i] = (n < N) ? f2bf(W[(size_t)k * N + n]) : (unsigned short)0;
}

// ---------------- WMMA GEMM: C[M x N] = A[M x K] * Bt^T (+bias, relu) -------
// One wave computes a 16-row M tile across all NT 16-col N tiles.
// A: row-major f32 (K mult of 32, 16B-aligned rows). Bt: col-major bf16.
// All fragments loaded straight from global per the ISA VGPR layouts:
//   A lane(l): row=lm, K in [8*lg,+8) u [16+8*lg,+8)  -> 4x float4
//   B lane(l): col=lm, K in [8*lg,+8) u [16+8*lg,+8)  -> 2x b128
//   C vgpr r : row = r + 8*lg, col = lm
template <int NT, int K>
__global__ void __launch_bounds__(256)
wmma_gemm_kernel(const float* __restrict__ A,
                 const unsigned short* __restrict__ Bt,
                 const float* __restrict__ bias, float* __restrict__ C,
                 int mtiles, int N, int applyBiasRelu)
{
  const int wave = (blockIdx.x * blockDim.x + threadIdx.x) >> 5;
  const int lane = threadIdx.x & 31;
  if (wave >= mtiles) return;                 // wave-uniform: EXEC stays full
  const int m0 = wave * 16;
  const int lg = lane >> 4;
  const int lm = lane & 15;

  const float* arow = A + (size_t)(m0 + lm) * K;

  v8f c[NT];
#pragma unroll
  for (int t = 0; t < NT; ++t)
#pragma unroll
    for (int r = 0; r < 8; ++r) c[t][r] = 0.0f;

#pragma unroll
  for (int k0 = 0; k0 < K; k0 += 32) {
    // A fragment: 16 floats -> bf16 in registers
    const float4* ap0 = (const float4*)(arow + k0 + 8 * lg);
    const float4* ap1 = (const float4*)(arow + k0 + 16 + 8 * lg);
    float4 a0 = ap0[0], a1 = ap0[1];
    float4 a2 = ap1[0], a3 = ap1[1];
    union { v16bf v; unsigned short u[16]; } fa;
    float af[16] = {a0.x, a0.y, a0.z, a0.w, a1.x, a1.y, a1.z, a1.w,
                    a2.x, a2.y, a2.z, a2.w, a3.x, a3.y, a3.z, a3.w};
#pragma unroll
    for (int j = 0; j < 16; ++j) fa.u[j] = f2bf(af[j]);

#pragma unroll
    for (int t = 0; t < NT; ++t) {
      union { v16bf v; uint4 q[2]; } fb;
      const uint4* bp =
          (const uint4*)(Bt + (size_t)(t * 16 + lm) * K + k0 + 8 * lg);
      fb.q[0] = bp[0];
      fb.q[1] = bp[2];
      c[t] = __builtin_amdgcn_wmma_f32_16x16x32_bf16(false, fa.v, false, fb.v,
                                                     (short)0, c[t], false, false);
    }
  }

#pragma unroll
  for (int t = 0; t < NT; ++t) {
    int col = t * 16 + lm;
    if (col < N) {
      float b = applyBiasRelu ? bias[col] : 0.0f;
#pragma unroll
      for (int r = 0; r < 8; ++r) {
        int row = m0 + r + 8 * lg;
        float v = c[t][r] + b;
        if (applyBiasRelu) v = v > 0.0f ? v : 0.0f;
        C[(size_t)row * N + col] = v;
      }
    }
  }
}

// ---------------- attention scalars: a_s[i] = h[i,:].att_src, a_d likewise ---
__global__ void att_dots_kernel(const float* __restrict__ h,
                                const float* __restrict__ att_s,
                                const float* __restrict__ att_d,
                                float* __restrict__ as_, float* __restrict__ ad_,
                                int n)
{
  int node = blockIdx.x * (blockDim.x >> 5) + (threadIdx.x >> 5);
  int lane = threadIdx.x & 31;
  if (node >= n) return;
  float2 hv = ((const float2*)(h + (size_t)node * HID))[lane];
  float2 sv = ((const float2*)att_s)[lane];
  float2 dv = ((const float2*)att_d)[lane];
  float s = hv.x * sv.x + hv.y * sv.y;
  float d = hv.x * dv.x + hv.y * dv.y;
#pragma unroll
  for (int o = 16; o > 0; o >>= 1) {
    s += __shfl_down(s, o, 32);
    d += __shfl_down(d, o, 32);
  }
  if (lane == 0) { as_[node] = s; ad_[node] = d; }
}

__global__ void fill_kernel(float* __restrict__ p, float v, int n) {
  int i = blockIdx.x * blockDim.x + threadIdx.x;
  if (i < n) p[i] = v;
}

// in-degree counts (incl. self loops); shared across both layers
__global__ void count_kernel(const int* __restrict__ dst, float* __restrict__ cnt,
                             int E, int n)
{
  int e = blockIdx.x * blockDim.x + threadIdx.x;
  if (e >= E + n) return;
  int d = (e < E) ? dst[e] : (e - E);
  atomicAdd(&cnt[d], 1.0f);
}

// float atomic max via signed/unsigned integer atomics (valid for IEEE order)
__device__ __forceinline__ void atomicMaxF(float* addr, float val) {
  if (val >= 0.0f) atomicMax((int*)addr, __float_as_int(val));
  else             atomicMin((unsigned int*)addr, __float_as_uint(val));
}

// pass 1: e = leakyrelu(a_s[src] + a_d[dst]); segment max over dst
__global__ void edge_score_max_kernel(const int* __restrict__ src,
                                      const int* __restrict__ dst,
                                      const float* __restrict__ as_,
                                      const float* __restrict__ ad_,
                                      float* __restrict__ eval_,
                                      float* __restrict__ mmax, int E, int n)
{
  int e = blockIdx.x * blockDim.x + threadIdx.x;
  if (e >= E + n) return;
  int s = (e < E) ? src[e] : (e - E);
  int d = (e < E) ? dst[e] : (e - E);
  float v = as_[s] + ad_[d];
  v = (v > 0.0f) ? v : NEG_SLOPE * v;
  eval_[e] = v;
  atomicMaxF(&mmax[d], v);
}

// pass 2: ex = exp(e - m[dst]); denom[dst] += ex; acc[dst,:] += ex * h[src,:]
// one wave32 per edge; each lane covers 2 adjacent channels (b64 load)
__global__ void edge_accum_kernel(const int* __restrict__ src,
                                  const int* __restrict__ dst,
                                  const float* __restrict__ eval_,
                                  const float* __restrict__ mmax,
                                  const float* __restrict__ h,
                                  float* __restrict__ denom,
                                  float* __restrict__ acc, int E, int n)
{
  int e    = blockIdx.x * (blockDim.x >> 5) + (threadIdx.x >> 5);
  int lane = threadIdx.x & 31;
  if (e >= E + n) return;
  int s = (e < E) ? src[e] : (e - E);
  int d = (e < E) ? dst[e] : (e - E);
  float ex = __expf(eval_[e] - mmax[d]);
  if (lane == 0) atomicAdd(&denom[d], ex);
  float2 hv = ((const float2*)(h + (size_t)s * HID))[lane];
  float* ad = acc + (size_t)d * HID + 2 * lane;
  atomicAdd(ad,     ex * hv.x);
  atomicAdd(ad + 1, ex * hv.y);
}

// in-place: acc = relu(acc / (denom * cnt) + bias)
__global__ void finalize_kernel(float* __restrict__ acc,
                                const float* __restrict__ denom,
                                const float* __restrict__ cnt,
                                const float* __restrict__ bias, int n)
{
  int i = blockIdx.x * blockDim.x + threadIdx.x;
  if (i >= n * HID) return;
  int node = i >> 6;
  int j    = i & (HID - 1);
  float v = acc[i] / (denom[node] * cnt[node]) + bias[j];
  acc[i] = v > 0.0f ? v : 0.0f;
}

static inline int cdiv(int a, int b) { return (a + b - 1) / b; }

extern "C" void kernel_launch(void* const* d_in, const int* in_sizes, int n_in,
                              void* d_out, int out_size, void* d_ws, size_t ws_size,
                              hipStream_t stream)
{
  const float* x      = (const float*)d_in[0];
  const int*   eidx   = (const int*)d_in[1];
  const float* W1     = (const float*)d_in[2];
  const float* att_s1 = (const float*)d_in[3];
  const float* att_d1 = (const float*)d_in[4];
  const float* bias1  = (const float*)d_in[5];
  const float* W2     = (const float*)d_in[6];
  const float* att_s2 = (const float*)d_in[7];
  const float* att_d2 = (const float*)d_in[8];
  const float* bias2  = (const float*)d_in[9];
  const float* lin_w  = (const float*)d_in[10];
  const float* lin_b  = (const float*)d_in[11];
  float* out = (float*)d_out;

  const int IN_DIM  = 128;
  const int N       = in_sizes[0] / IN_DIM;   // 50000
  const int E       = in_sizes[1] / 2;        // 800000
  const int OUT_DIM = in_sizes[11];           // 40
  const int* srcp = eidx;
  const int* dstp = eidx + E;
  const int ET = E + N;                       // with self loops

  // workspace layout: bf16 weight buffer first (16B-aligned), then f32 pool
  unsigned short* btbuf = (unsigned short*)d_ws;        // 64*128 = 8192 elems
  float* ws    = (float*)((char*)d_ws + 8192 * sizeof(unsigned short));
  float* buf0  = ws; ws += (size_t)N * HID;   // h_pre (GEMM output)
  float* buf1  = ws; ws += (size_t)N * HID;   // acc / layer output
  float* as_   = ws; ws += N;
  float* ad_   = ws; ws += N;
  float* mmax  = ws; ws += N;
  float* denom = ws; ws += N;
  float* cnt   = ws; ws += N;
  float* evalb = ws; ws += ET;

  const int mtiles = N / 16;                  // 3125
  const int gblk   = cdiv(mtiles * 32, 256);  // 8 waves (M tiles) per block

  // degree counts (shared by both layers)
  fill_kernel<<<cdiv(N, 256), 256, 0, stream>>>(cnt, 0.0f, N);
  count_kernel<<<cdiv(ET, 256), 256, 0, stream>>>(dstp, cnt, E, N);

  const float* ats[2] = {att_s1, att_s2};
  const float* atd[2] = {att_d1, att_d2};
  const float* bs[2]  = {bias1, bias2};

  for (int L = 0; L < 2; ++L) {
    // h_pre = layer_in @ W  (bf16 WMMA, f32 accumulate)
    if (L == 0) {
      conv_wt_kernel<<<cdiv(64 * 128, 256), 256, 0, stream>>>(W1, btbuf,
                                                              128, HID, 64);
      wmma_gemm_kernel<4, 128><<<gblk, 256, 0, stream>>>(x, btbuf, nullptr,
                                                         buf0, mtiles, HID, 0);
    } else {
      conv_wt_kernel<<<cdiv(64 * 64, 256), 256, 0, stream>>>(W2, btbuf,
                                                             64, HID, 64);
      wmma_gemm_kernel<4, 64><<<gblk, 256, 0, stream>>>(buf1, btbuf, nullptr,
                                                        buf0, mtiles, HID, 0);
    }
    // attention scalars
    att_dots_kernel<<<cdiv(N, 8), 256, 0, stream>>>(buf0, ats[L], atd[L],
                                                    as_, ad_, N);
    // init per-layer accumulators
    fill_kernel<<<cdiv(N, 256), 256, 0, stream>>>(mmax, -INFINITY, N);
    fill_kernel<<<cdiv(N, 256), 256, 0, stream>>>(denom, 0.0f, N);
    fill_kernel<<<cdiv(N * HID, 256), 256, 0, stream>>>(buf1, 0.0f, N * HID);
    // edge passes
    edge_score_max_kernel<<<cdiv(ET, 256), 256, 0, stream>>>(
        srcp, dstp, as_, ad_, evalb, mmax, E, N);
    edge_accum_kernel<<<cdiv(ET, 8), 256, 0, stream>>>(
        srcp, dstp, evalb, mmax, buf0, denom, buf1, E, N);
    // buf1 = relu(buf1 / (denom*cnt) + bias)
    finalize_kernel<<<cdiv(N * HID, 256), 256, 0, stream>>>(buf1, denom, cnt,
                                                            bs[L], N);
  }

  // out = relu(buf1 @ lin_w + lin_b): N=40 -> 3 N-tiles, masked store
  conv_wt_kernel<<<cdiv(48 * 64, 256), 256, 0, stream>>>(lin_w, btbuf,
                                                         64, OUT_DIM, 48);
  wmma_gemm_kernel<3, 64><<<gblk, 256, 0, stream>>>(buf1, btbuf, lin_b,
                                                    out, mtiles, OUT_DIM, 1);
}
